// LearnNMSModule_50680614093491
// MI455X (gfx1250) — matrix operand order repaired and checked by reading, not compiled, not run on hardware
//
#include <hip/hip_runtime.h>
#include <hip/hip_bf16.h>

// ---------------- problem constants (from reference shapes) ----------------
#define BATCH   2
#define NBOX    1000
#define NCLS    80
#define F_N     100      // FIRST_N
#define F_PAD   112      // F_N padded to multiple of 16 (zero rows)
#define FEAT    1024
#define DMODEL  128
#define EDIM    64
#define GRP     16
#define ATTD    1024
#define DK      64       // ATTD / GRP
#define NT      3
#define FP      112      // padded F inside attention
#define KP      128      // padded K for the w@fc GEMM
#define SCALE_CLAMP 4.135166556742356f  // log(1000/16)

typedef __attribute__((ext_vector_type(16))) _Float16 v16h;
typedef __attribute__((ext_vector_type(8)))  _Float16 v8h;
typedef __attribute__((ext_vector_type(8)))  float    v8f;

// ---------------- WMMA fragment helpers (CDNA5 v_wmma_f32_16x16x32_f16) ----
// A (16x32 f16): lane<16 -> M=lane, K = {0..7}+{16..23}; lane>=16 -> M=lane-16,
//                K = {8..15}+{24..31}. Two contiguous 8-half chunks per lane.
// B (32x16 f16): lane<16 -> N=lane, K = {0..7}+{16..23}; lane>=16 -> N=lane-16,
//                K = {8..15}+{24..31}. Two contiguous 8-half chunks per lane of B^T rows.
__device__ __forceinline__ v8f wmma32(v16h a, v16h b, v8f c) {
  return __builtin_amdgcn_wmma_f32_16x16x32_f16(false, a, false, b, (short)0, c,
                                                false, false);
}

__device__ __forceinline__ v16h frag_a(const _Float16* A, size_t lda, int rb, int kb) {
  int lane = threadIdx.x & 31;
  int m  = rb + (lane & 15);
  int kh = (lane < 16) ? 0 : 8;
  const _Float16* p = A + (size_t)m * lda + kb + kh;
  v8h lo = *(const v8h*)p;
  v8h hi = *(const v8h*)(p + 16);
  return __builtin_shufflevector(lo, hi, 0,1,2,3,4,5,6,7,8,9,10,11,12,13,14,15);
}

__device__ __forceinline__ v16h frag_b(const _Float16* Bt, size_t ldb, int nb, int kb) {
  int lane = threadIdx.x & 31;
  int n  = nb + (lane & 15);
  int ko = (lane < 16) ? 0 : 8;
  const _Float16* p = Bt + (size_t)n * ldb + kb + ko;
  v8h lo = *(const v8h*)p;
  v8h hi = *(const v8h*)(p + 16);
  return __builtin_shufflevector(lo, hi, 0,1,2,3,4,5,6,7,8,9,10,11,12,13,14,15);
}

// ---------------- generic f16 GEMM: C[M,N] = A[M,K] * Bt[N,K]^T + bias ------
// REQUIRES: M % 16 == 0, N % 16 == 0, K % 64 == 0 (callers pad).
// block = 128 threads (4 waves); wave handles N-tile (blockIdx.y*4 + wave).
// No per-lane guards -> no exec-mask churn in the K loop; K unrolled x2 so
// the next step's global loads overlap the current v_wmma.
__global__ void __launch_bounds__(128)
k_gemm_f16(const _Float16* __restrict__ A, const _Float16* __restrict__ Bt,
           const float* __restrict__ bias, float* __restrict__ Cf,
           _Float16* __restrict__ Ch, int M, int N, int K, int out_half) {
  int lane = threadIdx.x & 31;
  int wave = threadIdx.x >> 5;
  int rb = blockIdx.x * 16;
  int nb = (blockIdx.y * 4 + wave) * 16;
  v8f acc;
#pragma unroll
  for (int i = 0; i < 8; ++i) acc[i] = 0.0f;
  for (int kb = 0; kb < K; kb += 64) {
    v16h a0 = frag_a(A, K, rb, kb);
    v16h b0 = frag_b(Bt, K, nb, kb);
    v16h a1 = frag_a(A, K, rb, kb + 32);
    v16h b1 = frag_b(Bt, K, nb, kb + 32);
    acc = wmma32(a0, b0, acc);
    acc = wmma32(a1, b1, acc);
  }
  int col  = nb + (lane & 15);
  int rofs = (lane < 16) ? 0 : 8;
  float bv = bias[col];
#pragma unroll
  for (int v = 0; v < 8; ++v) {
    int row = rb + rofs + v;
    float val = acc[v] + bv;
    if (out_half) Ch[(size_t)row * N + col] = (_Float16)val;
    else          Cf[(size_t)row * N + col] = val;
  }
  (void)M;
}

// ---------------- elementwise / prep kernels -------------------------------
__global__ void k_apply_deltas(const float* __restrict__ deltas,
                               const float* __restrict__ boxes,
                               float* __restrict__ refined) {
  int t = blockIdx.x * blockDim.x + threadIdx.x;
  if (t >= BATCH * NBOX * NCLS) return;
  int r = t % NCLS, i = t / NCLS;
  const float* bx = boxes + (size_t)i * 4;
  float w = bx[2] - bx[0], h = bx[3] - bx[1];
  float cx = bx[0] + 0.5f * w, cy = bx[1] + 0.5f * h;
  const float* d = deltas + (size_t)i * NCLS * 4 + r * 4;
  float dx = d[0] * 0.1f, dy = d[1] * 0.1f;
  float dw = fminf(d[2] * 0.2f, SCALE_CLAMP);
  float dh = fminf(d[3] * 0.2f, SCALE_CLAMP);
  float pcx = dx * w + cx, pcy = dy * h + cy;
  float pw = __expf(dw) * w, ph = __expf(dh) * h;
  float* o = refined + (size_t)i * NCLS * 4 + r * 4;
  o[0] = pcx - 0.5f * pw; o[1] = pcy - 0.5f * ph;
  o[2] = pcx + 0.5f * pw; o[3] = pcy + 0.5f * ph;
}

__global__ void k_f32_to_h(const float* __restrict__ s, _Float16* __restrict__ d, int n) {
  int t = blockIdx.x * blockDim.x + threadIdx.x;
  if (t < n) d[t] = (_Float16)s[t];
}

// src [K][N] f32 -> dst [N][K] f16 (pre-transpose so WMMA B-fragments are contiguous)
__global__ void k_transpose_h(const float* __restrict__ s, _Float16* __restrict__ d,
                              int K, int N) {
  int t = blockIdx.x * blockDim.x + threadIdx.x;
  if (t >= K * N) return;
  int n = t % N, k = t / N;
  d[(size_t)n * K + k] = (_Float16)s[(size_t)k * N + n];
}

// rank embedding, padded to F_PAD rows (rows >= F_N are zero so the GEMM
// needs no row guards)
__global__ void k_rank_emb(_Float16* __restrict__ out) {
  int t = blockIdx.x * blockDim.x + threadIdx.x;
  if (t >= F_PAD * FEAT) return;
  int i = t % FEAT, r = t / FEAT;
  float v = 0.0f;
  if (r < F_N) {
    int fr = (i < FEAT / 2) ? i : i - FEAT / 2;
    float dim = __powf(1000.0f, (float)fr * (2.0f / (float)FEAT));
    float div = (float)r / dim;
    v = (i < FEAT / 2) ? __sinf(div) : __cosf(div);
  }
  out[t] = (_Float16)v;
}

// softmax over the 1000 boxes, one block per (b, class); probs [B][C][NB]
__global__ void __launch_bounds__(256)
k_softmax(const float* __restrict__ scores, float* __restrict__ probs) {
  __shared__ float red[256];
  int b = blockIdx.x / NCLS, c = blockIdx.x % NCLS;
  int tid = threadIdx.x;
  float mx = -1e30f;
  for (int i = tid; i < NBOX; i += 256)
    mx = fmaxf(mx, scores[(size_t)(b * NBOX + i) * (NCLS + 1) + c]);
  red[tid] = mx; __syncthreads();
  for (int s = 128; s > 0; s >>= 1) {
    if (tid < s) red[tid] = fmaxf(red[tid], red[tid + s]);
    __syncthreads();
  }
  mx = red[0]; __syncthreads();
  float sum = 0.0f;
  for (int i = tid; i < NBOX; i += 256)
    sum += __expf(scores[(size_t)(b * NBOX + i) * (NCLS + 1) + c] - mx);
  red[tid] = sum; __syncthreads();
  for (int s = 128; s > 0; s >>= 1) {
    if (tid < s) red[tid] += red[tid + s];
    __syncthreads();
  }
  float inv = 1.0f / red[0];
  for (int i = tid; i < NBOX; i += 256)
    probs[(size_t)(b * NCLS + c) * NBOX + i] =
        __expf(scores[(size_t)(b * NBOX + i) * (NCLS + 1) + c] - mx) * inv;
}

// stable top-100 selection per (b,c): repeated argmax, ties -> lowest index
__global__ void __launch_bounds__(256)
k_topk(const float* __restrict__ probs, int* __restrict__ order,
       float* __restrict__ outScore) {
  __shared__ float vals[NBOX];
  __shared__ float rv[256];
  __shared__ int   ri[256];
  int b = blockIdx.x / NCLS, c = blockIdx.x % NCLS;
  int tid = threadIdx.x;
  const float* p = probs + (size_t)(b * NCLS + c) * NBOX;
  for (int i = tid; i < NBOX; i += 256) vals[i] = p[i];
  __syncthreads();
  for (int t = 0; t < F_N; ++t) {
    float bv = -1e30f; int bi = NBOX;
    for (int i = tid; i < NBOX; i += 256) {
      float v = vals[i];
      if (v > bv) { bv = v; bi = i; }
    }
    rv[tid] = bv; ri[tid] = bi;
    __syncthreads();
    for (int s = 128; s > 0; s >>= 1) {
      if (tid < s) {
        if (rv[tid + s] > rv[tid] ||
            (rv[tid + s] == rv[tid] && ri[tid + s] < ri[tid])) {
          rv[tid] = rv[tid + s]; ri[tid] = ri[tid + s];
        }
      }
      __syncthreads();
    }
    if (tid == 0) {
      order[(b * F_N + t) * NCLS + c]    = ri[0];
      outScore[(b * F_N + t) * NCLS + c] = rv[0];
      vals[ri[0]] = -1e30f;
    }
    __syncthreads();
  }
}

// gather class-diagonal boxes through the torch .view(N,4,R) scrambling
__global__ void k_gather_bbox(const int* __restrict__ order,
                              const float* __restrict__ refined,
                              float* __restrict__ outBbox,
                              float* __restrict__ bboxcf) {
  int t = blockIdx.x * blockDim.x + threadIdx.x;
  if (t >= BATCH * F_N * NCLS) return;
  int c = t % NCLS, f = (t / NCLS) % F_N, b = t / (NCLS * F_N);
  int i = order[t];
  const float* src = refined + (size_t)(b * NBOX + i) * (NCLS * 4);
#pragma unroll
  for (int a = 0; a < 4; ++a) {
    float v = src[a * NCLS + c];           // column a*R + c of flat (r*4+coord) layout
    outBbox[(size_t)t * 4 + a] = v;
    bboxcf[((size_t)(b * NCLS + c) * F_N + f) * 4 + a] = v;
  }
}

// emb[b,f,c,:] = rank_proj[f] + roi_proj[rank_idx]; also f16 copy in [B,C,F,D]
__global__ void k_build_emb(const float* __restrict__ rankp, const float* __restrict__ roip,
                            const int* __restrict__ order, float* __restrict__ embf,
                            _Float16* __restrict__ embh) {
  int t = blockIdx.x * blockDim.x + threadIdx.x;
  if (t >= BATCH * F_N * NCLS * DMODEL) return;
  int d = t % DMODEL, rest = t / DMODEL;
  int c = rest % NCLS, f = (rest / NCLS) % F_N, b = rest / (NCLS * F_N);
  int i = order[rest];
  float v = rankp[f * DMODEL + d] + roip[(size_t)(b * NBOX + i) * DMODEL + d];
  embf[t] = v;
  embh[((size_t)(b * NCLS + c) * F_N + f) * DMODEL + d] = (_Float16)v;
}

// fused position-embedding -> log(relu(pos_emb @ W_pos + b_pos)) per head
__global__ void __launch_bounds__(256)
k_laff(const float* __restrict__ bboxcf, const float* __restrict__ Wpos,
       const float* __restrict__ bpos, _Float16* __restrict__ laff) {
  __shared__ float wp[EDIM * GRP + GRP];
  for (int i = threadIdx.x; i < EDIM * GRP; i += 256) wp[i] = Wpos[i];
  for (int i = threadIdx.x; i < GRP; i += 256) wp[EDIM * GRP + i] = bpos[i];
  __syncthreads();
  int t = blockIdx.x * blockDim.x + threadIdx.x;
  if (t >= BATCH * NCLS * F_N * F_N) return;
  int m = t % F_N, n = (t / F_N) % F_N, bc = t / (F_N * F_N);
  const float* bn = bboxcf + (size_t)(bc * F_N + n) * 4;
  const float* bm = bboxcf + (size_t)(bc * F_N + m) * 4;
  float cxn = 0.5f * (bn[0] + bn[2]), cyn = 0.5f * (bn[1] + bn[3]);
  float wn = bn[2] - bn[0] + 1.0f,    hn = bn[3] - bn[1] + 1.0f;
  float cxm = 0.5f * (bm[0] + bm[2]), cym = 0.5f * (bm[1] + bm[3]);
  float wm = bm[2] - bm[0] + 1.0f,    hm = bm[3] - bm[1] + 1.0f;
  float pm4[4];
  pm4[0] = __logf(fmaxf(fabsf(cxn - cxm) / wn, 1e-3f));
  pm4[1] = __logf(fmaxf(fabsf(cyn - cym) / hn, 1e-3f));
  pm4[2] = __logf(wn / wm);
  pm4[3] = __logf(hn / hm);
  const float dimv[8] = {1.0f, 2.3713737f, 5.6234133f, 13.335215f,
                         31.622776f, 74.989418f, 177.82794f, 421.69650f};
  float acc[GRP];
#pragma unroll
  for (int g = 0; g < GRP; ++g) acc[g] = wp[EDIM * GRP + g];
#pragma unroll
  for (int j = 0; j < 4; ++j) {
#pragma unroll
    for (int fq = 0; fq < 8; ++fq) {
      float div = 100.0f * pm4[j] / dimv[fq];
      float sv, cv; __sincosf(div, &sv, &cv);
      int is = j * 16 + fq, ic = j * 16 + 8 + fq;
#pragma unroll
      for (int g = 0; g < GRP; ++g)
        acc[g] += sv * wp[is * GRP + g] + cv * wp[ic * GRP + g];
    }
  }
  size_t nm = (size_t)n * F_N + m;
#pragma unroll
  for (int g = 0; g < GRP; ++g) {
    float a = fmaxf(acc[g], 0.0f);
    a = __logf(fmaxf(a, 1e-6f));
    laff[((size_t)(bc * GRP + g)) * (F_N * F_N) + nm] = (_Float16)a;
  }
}

// ---------------- fused attention: one block per (b,c,g) -------------------
__global__ void __launch_bounds__(256)
k_attention(const _Float16* __restrict__ qg, const _Float16* __restrict__ kg,
            const _Float16* __restrict__ embh, const _Float16* __restrict__ laff,
            const float* __restrict__ Wout, const float* __restrict__ bout,
            float* __restrict__ attout) {
  extern __shared__ char smem[];
  _Float16* qs  = (_Float16*)smem;            // [FP][DK]
  _Float16* ks_ = qs  + FP * DK;              // [FP][DK]
  _Float16* fcT = ks_ + FP * DK;              // [DMODEL][KP]  (fc transposed, K padded)
  _Float16* wH  = fcT + DMODEL * KP;          // [FP][KP]
  float*    fbuf = (float*)(wH + FP * KP);    // FP*DMODEL floats: aff then attg

  int tid = threadIdx.x, lane = tid & 31, wave = tid >> 5;
  int g = blockIdx.x & 15;
  int c = (blockIdx.x >> 4) % NCLS;
  int b = (blockIdx.x >> 4) / NCLS;
  int bc = b * NCLS + c;

  // stage 1: stage q/k head slices + transposed fc into LDS, zero pads
  for (int i = tid; i < FP * DK; i += 256) {
    int f = i / DK, d = i % DK;
    _Float16 qv = (_Float16)0.0f, kv = (_Float16)0.0f;
    if (f < F_N) {
      size_t idx = (size_t)(bc * F_N + f) * ATTD + g * DK + d;
      qv = qg[idx]; kv = kg[idx];
    }
    qs[i] = qv; ks_[i] = kv;
  }
  for (int i = tid; i < DMODEL * KP; i += 256) {
    int d = i / KP, m = i % KP;
    fcT[i] = (m < F_N) ? embh[(size_t)(bc * F_N + m) * DMODEL + d] : (_Float16)0.0f;
  }
  for (int i = tid; i < FP * KP; i += 256) wH[i] = (_Float16)0.0f;
  __syncthreads();

  // stage 2: aff = q*k^T/sqrt(dk) + logAff  (7x7 WMMA tiles, K=64)
  const _Float16* lbase = laff + (size_t)(bc * GRP + g) * (F_N * F_N);
  for (int t = wave; t < 49; t += 8) {
    int rt = t / 7, ct = t % 7;
    v8f acc;
#pragma unroll
    for (int i = 0; i < 8; ++i) acc[i] = 0.0f;
    acc = wmma32(frag_a(qs, DK, rt * 16, 0),  frag_b(ks_, DK, ct * 16, 0),  acc);
    acc = wmma32(frag_a(qs, DK, rt * 16, 32), frag_b(ks_, DK, ct * 16, 32), acc);
    int col  = ct * 16 + (lane & 15);
    int rofs = (lane < 16) ? 0 : 8;
    if (col < F_N) {
#pragma unroll
      for (int v = 0; v < 8; ++v) {
        int row = rt * 16 + rofs + v;
        float lv = (row < F_N) ? (float)lbase[row * F_N + col] : 0.0f;
        fbuf[row * FP + col] = acc[v] * 0.125f + lv;  // 1/sqrt(64)
      }
    }
  }
  __syncthreads();

  // stage 3: row softmax -> wH (f16), pads stay zero
  if (tid < F_N) {
    const float* rowp = fbuf + tid * FP;
    float mx = -1e30f;
    for (int m = 0; m < F_N; ++m) mx = fmaxf(mx, rowp[m]);
    float sum = 0.0f;
    for (int m = 0; m < F_N; ++m) sum += __expf(rowp[m] - mx);
    float inv = 1.0f / sum;
    for (int m = 0; m < F_N; ++m)
      wH[tid * KP + m] = (_Float16)(__expf(rowp[m] - mx) * inv);
  }
  __syncthreads();

  // stage 4: attg = w @ fc  (7x8 WMMA tiles, K padded to 128) -> fbuf [FP][DMODEL]
  for (int t = wave; t < 56; t += 8) {
    int rt = t / 8, dt = t % 8;
    v8f acc;
#pragma unroll
    for (int i = 0; i < 8; ++i) acc[i] = 0.0f;
#pragma unroll
    for (int kb = 0; kb < KP; kb += 32)
      acc = wmma32(frag_a(wH, KP, rt * 16, kb), frag_b(fcT, KP, dt * 16, kb), acc);
    int col  = dt * 16 + (lane & 15);
    int rofs = (lane < 16) ? 0 : 8;
#pragma unroll
    for (int v = 0; v < 8; ++v)
      fbuf[(rt * 16 + rofs + v) * DMODEL + col] = acc[v];
  }
  __syncthreads();

  // stage 5: grouped out-projection attg[100,128] @ W_out[g][128,8] + b_out
  const float* Wg = Wout + (size_t)g * DMODEL * (DMODEL / GRP);
  for (int i = tid; i < F_N * (DMODEL / GRP); i += 256) {
    int n = i / (DMODEL / GRP), e = i % (DMODEL / GRP);
    float acc = bout[g * (DMODEL / GRP) + e];
    for (int d = 0; d < DMODEL; ++d)
      acc += fbuf[n * DMODEL + d] * Wg[d * (DMODEL / GRP) + e];
    attout[(size_t)((b * F_N + n) * NCLS + c) * DMODEL + g * (DMODEL / GRP) + e] = acc;
  }
}

// relu(emb+att) @ W_logit -> sigmoid -> * sorted_score
__global__ void k_final(const float* __restrict__ embf, const float* __restrict__ attout,
                        const float* __restrict__ Wlog, const float* __restrict__ blog,
                        const float* __restrict__ sscore, float* __restrict__ outMulti) {
  int t = blockIdx.x * blockDim.x + threadIdx.x;
  if (t >= BATCH * F_N * NCLS) return;
  const float* e = embf   + (size_t)t * DMODEL;
  const float* a = attout + (size_t)t * DMODEL;
  float acc[NT];
#pragma unroll
  for (int nt = 0; nt < NT; ++nt) acc[nt] = blog[nt];
  for (int d = 0; d < DMODEL; ++d) {
    float af = fmaxf(e[d] + a[d], 0.0f);
#pragma unroll
    for (int nt = 0; nt < NT; ++nt) acc[nt] += af * Wlog[d * NT + nt];
  }
  float ss = sscore[t];
#pragma unroll
  for (int nt = 0; nt < NT; ++nt)
    outMulti[(size_t)t * NT + nt] = ss / (1.0f + __expf(-acc[nt]));
}

// ---------------- host orchestration ---------------------------------------
extern "C" void kernel_launch(void* const* d_in, const int* in_sizes, int n_in,
                              void* d_out, int out_size, void* d_ws, size_t ws_size,
                              hipStream_t stream) {
  (void)in_sizes; (void)n_in; (void)out_size; (void)ws_size;
  const float* roi_feat = (const float*)d_in[0];
  const float* scores   = (const float*)d_in[1];
  const float* deltas   = (const float*)d_in[2];
  const float* boxes    = (const float*)d_in[3];
  const float* W_roi    = (const float*)d_in[4];
  const float* b_roi    = (const float*)d_in[5];
  const float* W_rank   = (const float*)d_in[6];
  const float* b_rank   = (const float*)d_in[7];
  const float* W_logit  = (const float*)d_in[8];
  const float* b_logit  = (const float*)d_in[9];
  const float* W_pos    = (const float*)d_in[10];
  const float* b_pos    = (const float*)d_in[11];
  const float* W_q      = (const float*)d_in[12];
  const float* b_q      = (const float*)d_in[13];
  const float* W_k      = (const float*)d_in[14];
  const float* b_k      = (const float*)d_in[15];
  const float* W_out    = (const float*)d_in[16];
  const float* b_out    = (const float*)d_in[17];

  float* out = (float*)d_out;
  float* outMulti = out;                                  // [B,F,C,NT] 48000
  float* outBbox  = out + BATCH * F_N * NCLS * NT;        // [B,F,C,4]  64000
  float* outScore = outBbox + BATCH * F_N * NCLS * 4;     // [B,F,C]    16000

  char* ws = (char*)d_ws;
  size_t off = 0;
  auto carve = [&](size_t bytes) -> char* {
    char* p = ws + off;
    off = (off + bytes + 255) & ~(size_t)255;
    return p;
  };
  float* refined = (float*)carve((size_t)BATCH * NBOX * NCLS * 4 * 4);
  float* probs   = (float*)carve((size_t)BATCH * NCLS * NBOX * 4);
  int*   order   = (int*)  carve((size_t)BATCH * F_N * NCLS * 4);
  float* bboxcf  = (float*)carve((size_t)BATCH * NCLS * F_N * 4 * 4);
  float* roip    = (float*)carve((size_t)BATCH * NBOX * DMODEL * 4);
  float* rankp   = (float*)carve((size_t)F_PAD * DMODEL * 4);
  float* embf    = (float*)carve((size_t)BATCH * F_N * NCLS * DMODEL * 4);
  float* attoutW = (float*)carve((size_t)BATCH * F_N * NCLS * DMODEL * 4);
  _Float16* roih    = (_Float16*)carve((size_t)BATCH * NBOX * FEAT * 2);
  _Float16* rankemb = (_Float16*)carve((size_t)F_PAD * FEAT * 2);
  _Float16* WtRoi   = (_Float16*)carve((size_t)DMODEL * FEAT * 2);
  _Float16* WtRank  = (_Float16*)carve((size_t)DMODEL * FEAT * 2);
  _Float16* WtQ     = (_Float16*)carve((size_t)ATTD * DMODEL * 2);
  _Float16* WtK     = (_Float16*)carve((size_t)ATTD * DMODEL * 2);
  _Float16* embh    = (_Float16*)carve((size_t)BATCH * NCLS * F_N * DMODEL * 2);
  _Float16* qh      = (_Float16*)carve((size_t)BATCH * NCLS * F_N * ATTD * 2);
  _Float16* kh      = (_Float16*)carve((size_t)BATCH * NCLS * F_N * ATTD * 2);
  _Float16* laff    = (_Float16*)carve((size_t)BATCH * NCLS * GRP * F_N * F_N * 2);

  // 1) box decode + precision prep
  k_apply_deltas<<<(BATCH * NBOX * NCLS + 255) / 256, 256, 0, stream>>>(deltas, boxes, refined);
  k_f32_to_h<<<(BATCH * NBOX * FEAT + 255) / 256, 256, 0, stream>>>(roi_feat, roih, BATCH * NBOX * FEAT);
  k_transpose_h<<<(FEAT * DMODEL + 255) / 256, 256, 0, stream>>>(W_roi, WtRoi, FEAT, DMODEL);
  k_transpose_h<<<(FEAT * DMODEL + 255) / 256, 256, 0, stream>>>(W_rank, WtRank, FEAT, DMODEL);
  k_transpose_h<<<(DMODEL * ATTD + 255) / 256, 256, 0, stream>>>(W_q, WtQ, DMODEL, ATTD);
  k_transpose_h<<<(DMODEL * ATTD + 255) / 256, 256, 0, stream>>>(W_k, WtK, DMODEL, ATTD);
  k_rank_emb<<<(F_PAD * FEAT + 255) / 256, 256, 0, stream>>>(rankemb);

  // 2) per-class softmax + stable top-100
  k_softmax<<<BATCH * NCLS, 256, 0, stream>>>(scores, probs);
  k_topk<<<BATCH * NCLS, 256, 0, stream>>>(probs, order, outScore);
  k_gather_bbox<<<(BATCH * F_N * NCLS + 255) / 256, 256, 0, stream>>>(order, refined, outBbox, bboxcf);

  // 3) WMMA GEMMs (all dims padded: M%16==0, N%16==0, K%64==0)
  { dim3 grid(125, 2); k_gemm_f16<<<grid, 128, 0, stream>>>(roih, WtRoi, b_roi, roip, ((_Float16*)nullptr), BATCH * NBOX, DMODEL, FEAT, 0); }
  { dim3 grid(7, 2);   k_gemm_f16<<<grid, 128, 0, stream>>>(rankemb, WtRank, b_rank, rankp, ((_Float16*)nullptr), F_PAD, DMODEL, FEAT, 0); }
  k_build_emb<<<(BATCH * F_N * NCLS * DMODEL + 255) / 256, 256, 0, stream>>>(rankp, roip, order, embf, embh);
  { dim3 grid(1000, 16);
    k_gemm_f16<<<grid, 128, 0, stream>>>(embh, WtQ, b_q, ((float*)nullptr), qh, BATCH * NCLS * F_N, ATTD, DMODEL, 1);
    k_gemm_f16<<<grid, 128, 0, stream>>>(embh, WtK, b_k, ((float*)nullptr), kh, BATCH * NCLS * F_N, ATTD, DMODEL, 1); }

  // 4) fused geometric log-affinity + fused attention (WMMA) + head
  k_laff<<<(BATCH * NCLS * F_N * F_N + 255) / 256, 256, 0, stream>>>(bboxcf, W_pos, b_pos, laff);
  size_t att_lds = (size_t)(FP * DK * 2 + DMODEL * KP + FP * KP) * sizeof(_Float16) +
                   (size_t)FP * DMODEL * sizeof(float);  // 147456 B
  k_attention<<<BATCH * NCLS * GRP, 256, att_lds, stream>>>(qh, kh, embh, laff, W_out, b_out, attoutW);
  k_final<<<(BATCH * F_N * NCLS + 255) / 256, 256, 0, stream>>>(embf, attoutW, W_logit, b_logit, outScore, outMulti);
}
